// ArcFaceLoss_37074157699736
// MI455X (gfx1250) — matrix operand order repaired and checked by reading, not compiled, not run on hardware
//
#include <hip/hip_runtime.h>
#include <math.h>

// ArcFace loss on MI455X: fused GEMM + online softmax.
// fp32 GEMM emulated with split-bf16 (hi+lo) on V_WMMA_F32_16X16X32_BF16:
//   x.w ~= xh.wh + xh.wl + xl.wh  (fp32 accum; error ~2^-17 relative)
// inputs: d_in[0]=x [2048,128] f32, d_in[1]=target [2048] int32,
//         d_in[2]=weight [100000,128] f32;  output: d_out[0] = mean NLL (f32)

#define EMBED   128
#define BATCH   2048
#define NCLASS  100000
#define CBLK    128                            // classes per block (8 waves x 16)
#define NCB     ((NCLASS + CBLK - 1) / CBLK)   // 782 class blocks
#define MBLK    32                             // batch rows per block (2 tiles/wave)
#define KC      64                             // K chunk staged in LDS
#define LDB     72                             // padded LDS stride (bf16 elements)

#define SCALE_F 70.0f
#define COS_M   0.8775825618903728f            // cos(0.5)
#define SIN_M   0.479425538604203f             // sin(0.5)
#define MM_C    0.2397127693021015f            // sin(pi-0.5)*0.5
#define NEG_INF (-__builtin_inff())

typedef __bf16 v16bf __attribute__((ext_vector_type(16)));
typedef __bf16 v8bf  __attribute__((ext_vector_type(8)));
typedef float  v8f   __attribute__((ext_vector_type(8)));
typedef unsigned short v4u16 __attribute__((ext_vector_type(4)));

__device__ __forceinline__ unsigned short f2bf_rn(float f) {
  unsigned u = __float_as_uint(f);
  u += 0x7fffu + ((u >> 16) & 1u);             // round to nearest even
  return (unsigned short)(u >> 16);
}
__device__ __forceinline__ float bf2f(unsigned short b) {
  return __uint_as_float((unsigned)b << 16);
}

#define CAT16(lo8, hi8) __builtin_shufflevector((lo8), (hi8), \
    0, 1, 2, 3, 4, 5, 6, 7, 8, 9, 10, 11, 12, 13, 14, 15)

#define WMMA_BF16(A, B, C) __builtin_amdgcn_wmma_f32_16x16x32_bf16( \
    false, (A), false, (B), (short)0, (C), false, false)

// ---- row L2-normalize + split into hi/lo bf16: one wave per 128-float row --
__global__ void rownorm_split_kernel(const float* __restrict__ in,
                                     unsigned short* __restrict__ hi,
                                     unsigned short* __restrict__ lo, int rows) {
  int row  = (int)((blockIdx.x * blockDim.x + threadIdx.x) >> 5);
  int lane = threadIdx.x & 31;
  if (row >= rows) return;
  float4 v = ((const float4*)(in + (size_t)row * EMBED))[lane];
  float ss = v.x * v.x + v.y * v.y + v.z * v.z + v.w * v.w;
  for (int m = 16; m >= 1; m >>= 1) ss += __shfl_xor(ss, m, 32);
  float inv = 1.0f / fmaxf(sqrtf(ss), 1e-12f);
  float f[4] = {v.x * inv, v.y * inv, v.z * inv, v.w * inv};
  v4u16 hv, lv;
#pragma unroll
  for (int j = 0; j < 4; ++j) {
    unsigned short h = f2bf_rn(f[j]);
    hv[j] = h;
    lv[j] = f2bf_rn(f[j] - bf2f(h));
  }
  *(v4u16*)(hi + (size_t)row * EMBED + lane * 4) = hv;
  *(v4u16*)(lo + (size_t)row * EMBED + lane * 4) = lv;
}

// ---- fused split-bf16 GEMM + margin epilogue + per-block online softmax ----
__global__ void __launch_bounds__(256)
arcface_gemm_kernel(const unsigned short* __restrict__ xh,
                    const unsigned short* __restrict__ xl,
                    const unsigned short* __restrict__ wh,
                    const unsigned short* __restrict__ wl,
                    const int* __restrict__ target,
                    float* __restrict__ pm, float* __restrict__ ps) {
  __shared__ __align__(16) __bf16 AsH[MBLK * LDB];   // 32 x 64 (padded)
  __shared__ __align__(16) __bf16 AsL[MBLK * LDB];
  __shared__ __align__(16) __bf16 BsH[CBLK * LDB];   // 128 x 64 (padded)
  __shared__ __align__(16) __bf16 BsL[CBLK * LDB];
  __shared__ float redM[8][MBLK];
  __shared__ float redS[8][MBLK];
  __shared__ int   tgt_s[MBLK];

  const int tid   = threadIdx.x;
  const int lane  = tid & 31;
  const int wave  = tid >> 5;
  const int l15   = lane & 15;
  const int half  = lane >> 4;                 // 0: lanes 0-15, 1: lanes 16-31
  const int cbase = blockIdx.x * CBLK;
  const int bbase = blockIdx.y * MBLK;

  if (tid < MBLK) tgt_s[tid] = target[bbase + tid];

  v8f acc0 = {0.f, 0.f, 0.f, 0.f, 0.f, 0.f, 0.f, 0.f};
  v8f acc1 = {0.f, 0.f, 0.f, 0.f, 0.f, 0.f, 0.f, 0.f};

  for (int kc = 0; kc < EMBED; kc += KC) {
    __syncthreads();
    { // stage A hi/lo: 32 rows x 32 bf16 = 128 uint4 per array
      int sel = tid >> 7;                      // 0 -> AsH, 1 -> AsL
      int t = tid & 127;
      int r = t >> 2, c = t & 3;
      const unsigned short* src = sel ? xl : xh;
      __bf16* dst = sel ? AsL : AsH;
      uint4 v = *(const uint4*)(src + (size_t)(bbase + r) * EMBED + kc + c * 8);
      *(uint4*)&dst[r * LDB + c * 8] = v;
    }
#pragma unroll
    for (int i = 0; i < 2; ++i) {              // stage B hi/lo: 512 uint4 each
      int idx = tid + i * 256;
      int r = idx >> 2, c = idx & 3;
      int g = cbase + r;
      uint4 vh = {0u, 0u, 0u, 0u}, vl = {0u, 0u, 0u, 0u};
      if (g < NCLASS) {
        vh = *(const uint4*)(wh + (size_t)g * EMBED + kc + c * 8);
        vl = *(const uint4*)(wl + (size_t)g * EMBED + kc + c * 8);
      }
      *(uint4*)&BsH[r * LDB + c * 8] = vh;
      *(uint4*)&BsL[r * LDB + c * 8] = vl;
    }
    __syncthreads();
#pragma unroll
    for (int kk = 0; kk < KC; kk += 32) {
      // A frag (16-bit 16x32 layout): lanes 0-15 carry K +0..7 / +16..23,
      // lanes 16-31 carry K +8..15 / +24..31
      int a0 = l15 * LDB + kk + 8 * half;
      int a1 = (l15 + 16) * LDB + kk + 8 * half;
      v16bf ah0 = CAT16(*(const v8bf*)&AsH[a0], *(const v8bf*)&AsH[a0 + 16]);
      v16bf al0 = CAT16(*(const v8bf*)&AsL[a0], *(const v8bf*)&AsL[a0 + 16]);
      v16bf ah1 = CAT16(*(const v8bf*)&AsH[a1], *(const v8bf*)&AsH[a1 + 16]);
      v16bf al1 = CAT16(*(const v8bf*)&AsL[a1], *(const v8bf*)&AsL[a1 + 16]);
      // B frag (32x16): column l15; lanes 0-15 carry K +0..7 / +16..23, lanes 16-31 K +8..15 / +24..31
      int bo = (wave * 16 + l15) * LDB + kk + 8 * half;
      v16bf bh = CAT16(*(const v8bf*)&BsH[bo], *(const v8bf*)&BsH[bo + 16]);
      v16bf bl = CAT16(*(const v8bf*)&BsL[bo], *(const v8bf*)&BsL[bo + 16]);
      // split-fp32 product: hi*hi + hi*lo + lo*hi
      acc0 = WMMA_BF16(ah0, bh, acc0);
      acc0 = WMMA_BF16(ah0, bl, acc0);
      acc0 = WMMA_BF16(al0, bh, acc0);
      acc1 = WMMA_BF16(ah1, bh, acc1);
      acc1 = WMMA_BF16(ah1, bl, acc1);
      acc1 = WMMA_BF16(al1, bh, acc1);
    }
  }

  // epilogue: margin transform + 16-lane (per-row) max / sumexp
  const int gclass = cbase + wave * 16 + l15;
#pragma unroll
  for (int t = 0; t < 2; ++t) {
    v8f acc = t ? acc1 : acc0;
#pragma unroll
    for (int r = 0; r < 8; ++r) {
      int m = r + half * 8 + t * 16;           // C layout: VGPR r -> M=r / r+8
      float v;
      if (gclass < NCLASS) {
        float c = fminf(fmaxf(acc[r], -1.f), 1.f);
        float s = sqrtf(fmaxf(1.f - c * c, 0.f));
        float adj = (gclass == tgt_s[m]) ? (c - MM_C) : (c * COS_M - s * SIN_M);
        v = adj * SCALE_F;
      } else {
        v = NEG_INF;                           // padding classes
      }
      float mx = v;
      mx = fmaxf(mx, __shfl_xor(mx, 1, 16));
      mx = fmaxf(mx, __shfl_xor(mx, 2, 16));
      mx = fmaxf(mx, __shfl_xor(mx, 4, 16));
      mx = fmaxf(mx, __shfl_xor(mx, 8, 16));
      float e = (v == NEG_INF) ? 0.f : expf(v - mx);
      e += __shfl_xor(e, 1, 16);
      e += __shfl_xor(e, 2, 16);
      e += __shfl_xor(e, 4, 16);
      e += __shfl_xor(e, 8, 16);
      if (l15 == 0) { redM[wave][m] = mx; redS[wave][m] = e; }
    }
  }
  __syncthreads();

  if (tid < MBLK) {                            // combine the 8 waves' (m,s)
    float m = NEG_INF, s = 0.f;
#pragma unroll
    for (int w = 0; w < 8; ++w) {
      float m2 = redM[w][tid], s2 = redS[w][tid];
      if (m2 != NEG_INF) {
        if (m == NEG_INF) { m = m2; s = s2; }
        else { float nm = fmaxf(m, m2);
               s = s * expf(m - nm) + s2 * expf(m2 - nm); m = nm; }
      }
    }
    pm[(size_t)blockIdx.x * BATCH + bbase + tid] = m;
    ps[(size_t)blockIdx.x * BATCH + bbase + tid] = s;
  }
}

// ------- per-row: combine class blocks, target logit, NLL (one wave/row) ----
__global__ void arcface_reduce_kernel(const float* __restrict__ pm,
                                      const float* __restrict__ ps,
                                      const unsigned short* __restrict__ xh,
                                      const unsigned short* __restrict__ xl,
                                      const unsigned short* __restrict__ wh,
                                      const unsigned short* __restrict__ wl,
                                      const int* __restrict__ target,
                                      float* __restrict__ nll) {
  int row  = (int)((blockIdx.x * blockDim.x + threadIdx.x) >> 5);
  int lane = threadIdx.x & 31;
  if (row >= BATCH) return;
  float m = NEG_INF, s = 0.f;
  for (int cb = lane; cb < NCB; cb += 32) {
    float m2 = pm[(size_t)cb * BATCH + row];
    float s2 = ps[(size_t)cb * BATCH + row];
    if (m2 != NEG_INF) {
      if (m == NEG_INF) { m = m2; s = s2; }
      else { float nm = fmaxf(m, m2);
             s = s * expf(m - nm) + s2 * expf(m2 - nm); m = nm; }
    }
  }
  for (int msk = 16; msk >= 1; msk >>= 1) {
    float m2 = __shfl_xor(m, msk, 32);
    float s2 = __shfl_xor(s, msk, 32);
    if (m2 != NEG_INF) {
      if (m == NEG_INF) { m = m2; s = s2; }
      else { float nm = fmaxf(m, m2);
             s = s * expf(m - nm) + s2 * expf(m2 - nm); m = nm; }
    }
  }
  int tgt = target[row];
  size_t xb = (size_t)row * EMBED, wb = (size_t)tgt * EMBED;
  float dot = 0.f;
  for (int i = lane; i < EMBED; i += 32) {
    float xa = bf2f(xh[xb + i]) + bf2f(xl[xb + i]);
    float wa = bf2f(wh[wb + i]) + bf2f(wl[wb + i]);
    dot += xa * wa;
  }
  for (int msk = 16; msk >= 1; msk >>= 1) dot += __shfl_xor(dot, msk, 32);
  if (lane == 0) {
    float c = fminf(fmaxf(dot, -1.f), 1.f);
    float lt = (c - MM_C) * SCALE_F;           // target position uses cos - mm
    nll[row] = m + logf(s) - lt;
  }
}

// ---------------- mean over 2048 NLL values ---------------------------------
__global__ void mean_kernel(const float* __restrict__ nll, float* __restrict__ out) {
  __shared__ float red[8];
  float s = 0.f;
  for (int i = threadIdx.x; i < BATCH; i += 256) s += nll[i];
  for (int msk = 16; msk >= 1; msk >>= 1) s += __shfl_xor(s, msk, 32);
  int lane = threadIdx.x & 31, wave = threadIdx.x >> 5;
  if (lane == 0) red[wave] = s;
  __syncthreads();
  if (threadIdx.x == 0) {
    float t = 0.f;
    for (int w = 0; w < 8; ++w) t += red[w];
    out[0] = t / (float)BATCH;
  }
}

extern "C" void kernel_launch(void* const* d_in, const int* in_sizes, int n_in,
                              void* d_out, int out_size, void* d_ws, size_t ws_size,
                              hipStream_t stream) {
  const float* x      = (const float*)d_in[0];
  const int*   target = (const int*)d_in[1];   // harness: integer -> int32
  const float* weight = (const float*)d_in[2];

  // workspace layout (~65 MB): fp32 partials first, then bf16 hi/lo splits
  float* pm  = (float*)d_ws;                              // NCB*BATCH
  float* ps  = pm + (size_t)NCB * BATCH;                  // NCB*BATCH
  float* nll = ps + (size_t)NCB * BATCH;                  // BATCH
  unsigned short* xh = (unsigned short*)(nll + BATCH);    // BATCH*EMBED
  unsigned short* xl = xh + (size_t)BATCH * EMBED;
  unsigned short* wh = xl + (size_t)BATCH * EMBED;        // NCLASS*EMBED
  unsigned short* wl = wh + (size_t)NCLASS * EMBED;

  rownorm_split_kernel<<<(BATCH * 32 + 255) / 256, 256, 0, stream>>>(
      x, xh, xl, BATCH);
  rownorm_split_kernel<<<(NCLASS * 32 + 255) / 256, 256, 0, stream>>>(
      weight, wh, wl, NCLASS);

  dim3 grid(NCB, BATCH / MBLK);                           // 782 x 64 blocks
  arcface_gemm_kernel<<<grid, 256, 0, stream>>>(xh, xl, wh, wl, target, pm, ps);

  arcface_reduce_kernel<<<(BATCH * 32 + 255) / 256, 256, 0, stream>>>(
      pm, ps, xh, xl, wh, wl, target, nll);
  mean_kernel<<<1, 256, 0, stream>>>(nll, (float*)d_out);
}
